// SGFormer_81947976007864
// MI455X (gfx1250) — matrix operand order repaired and checked by reading, not compiled, not run on hardware
//
#include <hip/hip_runtime.h>
#include <hip/hip_fp16.h>
#include <cstdint>

typedef __attribute__((ext_vector_type(16))) _Float16 v16h;
typedef __attribute__((ext_vector_type(8)))  float    v8f;

#define NN   30000
#define NNP  30016   /* NN padded to multiple of 32 */
#define NE   600000
#define NFLT 30000.0f

#define EPI_NONE      0
#define EPI_LN_RELU   1
#define EPI_TANH_NORM 2
#define EPI_RELU      3
#define EPI_NORM      4

union FragA { uint4 u[2]; v16h v; };

#define WMMA_F16(accv, av, bv) \
  __builtin_amdgcn_wmma_f32_16x16x32_f16(false, (av), false, (bv), (short)0, (accv), false, false)

__device__ __forceinline__ float redsum16(float v) {
  v += __shfl_xor(v, 1, 32);
  v += __shfl_xor(v, 2, 32);
  v += __shfl_xor(v, 4, 32);
  v += __shfl_xor(v, 8, 32);
  return v;
}

// CDNA5 async copy: global -> LDS, 16B per lane, tracked with ASYNCcnt.
__device__ __forceinline__ void async_copy16(const void* g, _Float16* lds) {
  __attribute__((address_space(3))) _Float16* l3 =
      (__attribute__((address_space(3))) _Float16*)lds;
  asm volatile("global_load_async_to_lds_b128 %0, %1, off"
               :: "v"(l3), "v"(g) : "memory");
}
__device__ __forceinline__ void async_wait0() {
  asm volatile("s_wait_asynccnt 0x0" ::: "memory");
}

// ---------------------------------------------------------------------------
// Pack f32 weights [Kin, Kout] into WMMA B-fragment order (f16):
// idx = ((ct*(Kin/32) + ks)*32 + lane)*16 + i  maps to  W[(ks*32 + hiB + i), ct*16 + (lane&15)]
// so a lane's 16 halves for one (ct, k-step) tile are 32 contiguous bytes.
// ---------------------------------------------------------------------------
__global__ void pack_w_kernel(const float* __restrict__ W, _Float16* __restrict__ Wp,
                              int Kin, int Kout) {
  const int total = Kin * Kout;
  const int kst = Kin >> 5;
  for (int idx = blockIdx.x * blockDim.x + threadIdx.x; idx < total;
       idx += blockDim.x * gridDim.x) {
    const int i    = idx & 15;
    const int lane = (idx >> 4) & 31;
    const int tw   = idx >> 9;
    const int ks   = tw % kst, ct = tw / kst;
    const int k = (ks << 5) + ((lane >> 4) << 4) + i;
    const int n = (ct << 4) + (lane & 15);
    Wp[idx] = (_Float16)W[(size_t)k * Kout + n];
  }
}

// kvs (f32 [2,256,256]) -> scaled f16 B-fragment layout per head.
__global__ void pack_kvs_kernel(const float* __restrict__ kvs32, const float* __restrict__ sqk,
                                _Float16* __restrict__ kvsp) {
  const float sc = rsqrtf(sqk[0]);
  for (int idx = blockIdx.x * blockDim.x + threadIdx.x; idx < 2 * 65536;
       idx += blockDim.x * gridDim.x) {
    const int h    = idx >> 16;
    const int rest = idx & 65535;
    const int i    = rest & 15;
    const int lane = (rest >> 4) & 31;
    const int tw   = rest >> 9;
    const int ks   = tw & 7, ct = tw >> 3;
    const int k = (ks << 5) + ((lane >> 4) << 4) + i;
    const int n = (ct << 4) + (lane & 15);
    kvsp[idx] = (_Float16)(kvs32[(size_t)h * 65536 + (size_t)k * 256 + n] * sc);
  }
}

// [NN,512] f32 -> [512,NNP] f16, zero-padded tail.
__global__ void transpose_cvt_kernel(const float* __restrict__ in, _Float16* __restrict__ out) {
  const size_t total = (size_t)512 * NNP;
  for (size_t idx = (size_t)blockIdx.x * blockDim.x + threadIdx.x; idx < total;
       idx += (size_t)gridDim.x * blockDim.x) {
    const int c = (int)(idx / NNP);
    const int n = (int)(idx - (size_t)c * NNP);
    out[idx] = (n < NN) ? (_Float16)in[(size_t)n * 512 + c] : (_Float16)0.0f;
  }
}

// ---------------------------------------------------------------------------
// Generic fused WMMA GEMM: 16 rows x full Kout per block, packed-f16 weights.
// ---------------------------------------------------------------------------
__launch_bounds__(256)
__global__ void gemm16(const _Float16* __restrict__ A, int lda, int a_coff,
                       const _Float16* __restrict__ Wp, const float* __restrict__ bias,
                       int Kin, int Kout, int epi,
                       const float* __restrict__ lng, const float* __restrict__ lnb,
                       float* __restrict__ out32,
                       _Float16* __restrict__ out16, int o16_ld, int o16_coff)
{
  __shared__ _Float16 As[16 * 512];
  __shared__ float    Os[16 * 512];
  const int tid  = threadIdx.x;
  const int row0 = blockIdx.x * 16;

  // Async-stage the 16 x Kin A-panel into LDS (16B chunks, ASYNCcnt-tracked).
  {
    const int rowChunks = Kin >> 3;          // 16B chunks per row
    const int total = 16 * rowChunks;
    for (int ch = tid; ch < total; ch += 256) {
      const int r = ch / rowChunks, cc = ch - r * rowChunks;
      async_copy16(A + (size_t)(row0 + r) * lda + a_coff + cc * 8,
                   As + r * Kin + cc * 8);
    }
    async_wait0();
  }
  __syncthreads();

  const int wave = tid >> 5, lane = tid & 31;
  const int m   = lane & 15;
  const int hiA = (lane >> 4) << 3;   // 0 or 8
  const int kst = Kin >> 5;
  const int ntiles = Kout >> 4;

  for (int ct = wave; ct < ntiles; ct += 8) {
    const int n0 = ct << 4;
    const _Float16* bbase = Wp + (size_t)ct * kst * 512 + lane * 16;
    __builtin_prefetch(bbase, 0, 3);
    v8f acc = {};
    for (int ks = 0; ks < kst; ++ks) {
      FragA a;
      const _Float16* abase = As + m * Kin + (ks << 5) + 2 * hiA;
      a.u[0] = *(const uint4*)(abase);
      a.u[1] = *(const uint4*)(abase + 8);
      FragA b;
      const _Float16* bp = bbase + (size_t)ks * 512;
      b.u[0] = *(const uint4*)(bp);
      b.u[1] = *(const uint4*)(bp + 8);
      acc = WMMA_F16(acc, a.v, b.v);
    }
    const int col = n0 + m;
    const float bv = bias ? bias[col] : 0.0f;
    #pragma unroll
    for (int r = 0; r < 8; ++r)
      Os[(r + hiA) * Kout + col] = acc[r] + bv;
  }
  __syncthreads();

  // Row-wise epilogue: 16 threads per row.
  const int row  = tid >> 4, t16 = tid & 15;
  const int node = row0 + row;
  float* orow = Os + row * Kout;

  if (epi == EPI_LN_RELU) {
    float s = 0.f, ss = 0.f;
    for (int c = t16; c < Kout; c += 16) { float v = orow[c]; s += v; ss += v * v; }
    s = redsum16(s); ss = redsum16(ss);
    const float inv  = 1.0f / (float)Kout;
    const float mu   = s * inv;
    const float var  = ss * inv - mu * mu;
    const float rstd = rsqrtf(var + 1e-5f);
    for (int c = t16; c < Kout; c += 16) {
      float v = (orow[c] - mu) * rstd * lng[c] + lnb[c];
      v = fmaxf(v, 0.0f);
      if (out32) out32[(size_t)node * Kout + c] = v;
      if (out16) out16[(size_t)node * o16_ld + o16_coff + c] = (_Float16)v;
    }
  } else if (epi == EPI_TANH_NORM) {
    float ss = 0.f;
    for (int c = t16; c < Kout; c += 16) { float v = tanhf(orow[c]); orow[c] = v; ss += v * v; }
    ss = redsum16(ss);
    const float sc = 1.0f / fmaxf(sqrtf(ss), 1e-12f);
    for (int c = t16; c < Kout; c += 16) {
      float v = orow[c] * sc;
      if (out32) out32[(size_t)node * Kout + c] = v;
      if (out16) out16[(size_t)node * o16_ld + o16_coff + c] = (_Float16)v;
    }
  } else if (epi == EPI_NORM) {
    float ss = 0.f;
    for (int c = t16; c < Kout; c += 16) { float v = orow[c]; ss += v * v; }
    ss = redsum16(ss);
    const float sc = 1.0f / fmaxf(sqrtf(ss), 1e-12f);
    for (int c = t16; c < Kout; c += 16) {
      float v = orow[c] * sc;
      if (out32) out32[(size_t)node * Kout + c] = v;
      if (out16) out16[(size_t)node * o16_ld + o16_coff + c] = (_Float16)v;
    }
  } else {
    for (int c = t16; c < Kout; c += 16) {
      float v = orow[c];
      if (epi == EPI_RELU) v = fmaxf(v, 0.0f);
      if (out32) out32[(size_t)node * Kout + c] = v;
      if (out16) out16[(size_t)node * o16_ld + o16_coff + c] = (_Float16)v;
    }
  }
}

// ---------------------------------------------------------------------------
// kvs[h] += K_h^T @ V_h : split-K WMMA over transposed f16 panels,
// LDS f32 reduce (ds_add_f32) + global_atomic_add_f32.
// grid: (512 = h*256 + mt*16 + dt, KCHUNK) x 256 threads
// ---------------------------------------------------------------------------
__launch_bounds__(256)
__global__ void kvs_kernel(const _Float16* __restrict__ kT, const _Float16* __restrict__ vT,
                           float* __restrict__ kvs)
{
  __shared__ float tile[256];
  const int tid = threadIdx.x;
  tile[tid] = 0.0f;
  __syncthreads();

  const int id = blockIdx.x;
  const int h = id >> 8, mt = (id >> 4) & 15, dt = id & 15;
  const int wave = tid >> 5, lane = tid & 31;
  const int m   = lane & 15;
  const int hiA = (lane >> 4) << 3;
  const int hiB = (lane >> 4) << 4;
  const _Float16* arow = kT + (size_t)(h * 256 + mt * 16 + m) * NNP;
  const _Float16* brow = vT + (size_t)(h * 256 + dt * 16 + m) * NNP;
  const int nsteps = NNP >> 5;              // 938
  const int wtot   = gridDim.y * 8;

  v8f acc = {};
  for (int s = blockIdx.y * 8 + wave; s < nsteps; s += wtot) {
    const int n0 = s << 5;
    FragA a, b;
    a.u[0] = *(const uint4*)(arow + n0 + hiB);
    a.u[1] = *(const uint4*)(arow + n0 + hiB + 8);
    b.u[0] = *(const uint4*)(brow + n0 + hiB);
    b.u[1] = *(const uint4*)(brow + n0 + hiB + 8);
    acc = WMMA_F16(acc, a.v, b.v);
  }
  #pragma unroll
  for (int r = 0; r < 8; ++r)
    atomicAdd(&tile[(r + hiA) * 16 + m], acc[r]);   // ds_add_f32
  __syncthreads();

  const int rr = tid >> 4, dd = tid & 15;
  unsafeAtomicAdd(&kvs[(size_t)(h * 256 + mt * 16 + rr) * 256 + dt * 16 + dd], tile[tid]);
}

// ---------------------------------------------------------------------------
// Fused attention tail: num = rq*(q @ kvs_scaled) + n*v ; attn = mean_h(num/denom)
// xt = 0.5*attn + 0.5*h0 ; comb[:, :256] = f16(relu(LN(xt)))
// ---------------------------------------------------------------------------
__launch_bounds__(256)
__global__ void attn_fuse(const _Float16* __restrict__ q16,
                          const _Float16* __restrict__ kvsp,
                          const float* __restrict__ vmat,
                          const float* __restrict__ h0,
                          const float* __restrict__ denom,
                          const float* __restrict__ sqq,
                          const float* __restrict__ lng, const float* __restrict__ lnb,
                          _Float16* __restrict__ comb)
{
  __shared__ _Float16 As[16 * 512];
  __shared__ float    Os[2][16 * 256];
  const int tid  = threadIdx.x;
  const int row0 = blockIdx.x * 16;

  for (int ch = tid; ch < 16 * 64; ch += 256) {     // 64 x 16B chunks per 512-half row
    const int r = ch >> 6, cc = ch & 63;
    async_copy16(q16 + (size_t)(row0 + r) * 512 + cc * 8, As + r * 512 + cc * 8);
  }
  async_wait0();
  __syncthreads();

  const int wave = tid >> 5, lane = tid & 31;
  const int m   = lane & 15;
  const int hiA = (lane >> 4) << 3;

  for (int h = 0; h < 2; ++h) {
    const _Float16* B = kvsp + (size_t)h * 65536;
    for (int ct = wave; ct < 16; ct += 8) {
      const int n0 = ct << 4;
      const _Float16* bbase = B + (size_t)ct * 8 * 512 + lane * 16;
      __builtin_prefetch(bbase, 0, 3);
      v8f acc = {};
      for (int ks = 0; ks < 8; ++ks) {
        FragA a;
        const _Float16* abase = As + m * 512 + h * 256 + (ks << 5) + 2 * hiA;
        a.u[0] = *(const uint4*)(abase);
        a.u[1] = *(const uint4*)(abase + 8);
        FragA b;
        const _Float16* bp = bbase + (size_t)ks * 512;
        b.u[0] = *(const uint4*)(bp);
        b.u[1] = *(const uint4*)(bp + 8);
        acc = WMMA_F16(acc, a.v, b.v);
      }
      const int col = n0 + m;
      #pragma unroll
      for (int r = 0; r < 8; ++r)
        Os[h][(r + hiA) * 256 + col] = acc[r];
    }
  }
  __syncthreads();

  const float rq = rsqrtf(sqq[0]);
  const int row  = tid >> 4, t16 = tid & 15;
  const int node = row0 + row;
  const float d0 = denom[(size_t)node * 2 + 0];
  const float d1 = denom[(size_t)node * 2 + 1];
  float* orow = Os[0] + row * 256;
  float* orw1 = Os[1] + row * 256;

  float s = 0.f, ss = 0.f;
  for (int c = t16; c < 256; c += 16) {
    float num0 = rq * orow[c] + NFLT * vmat[(size_t)node * 512 + c];
    float num1 = rq * orw1[c] + NFLT * vmat[(size_t)node * 512 + 256 + c];
    float attn = 0.5f * (num0 / d0 + num1 / d1);
    float xt = 0.5f * attn + 0.5f * h0[(size_t)node * 256 + c];
    orow[c] = xt; s += xt; ss += xt * xt;
  }
  s = redsum16(s); ss = redsum16(ss);
  const float mu   = s * (1.0f / 256.0f);
  const float var  = ss * (1.0f / 256.0f) - mu * mu;
  const float rstd = rsqrtf(var + 1e-5f);
  for (int c = t16; c < 256; c += 16) {
    float v = fmaxf((orow[c] - mu) * rstd * lng[c] + lnb[c], 0.0f);
    comb[(size_t)node * 512 + c] = (_Float16)v;
  }
}

// ------------------------- small helper kernels ----------------------------
__global__ void cvt_f16_kernel(const float* __restrict__ in, _Float16* __restrict__ out, int n) {
  for (int i = blockIdx.x * blockDim.x + threadIdx.x; i < n; i += blockDim.x * gridDim.x)
    out[i] = (_Float16)in[i];
}

__launch_bounds__(256)
__global__ void sumsq2_kernel(const float* __restrict__ q, const float* __restrict__ k,
                              float* __restrict__ sqq, float* __restrict__ sqk) {
  __shared__ float sa[256], sb[256];
  const size_t n = (size_t)NN * 512;
  float a = 0.f, b = 0.f;
  for (size_t i = (size_t)blockIdx.x * 256 + threadIdx.x; i < n; i += (size_t)gridDim.x * 256) {
    float x = q[i]; a += x * x;
    float y = k[i]; b += y * y;
  }
  sa[threadIdx.x] = a; sb[threadIdx.x] = b;
  __syncthreads();
  for (int s = 128; s > 0; s >>= 1) {
    if ((int)threadIdx.x < s) { sa[threadIdx.x] += sa[threadIdx.x + s]; sb[threadIdx.x] += sb[threadIdx.x + s]; }
    __syncthreads();
  }
  if (threadIdx.x == 0) { unsafeAtomicAdd(sqq, sa[0]); unsafeAtomicAdd(sqk, sb[0]); }
}

__global__ void colsum_kernel(const float* __restrict__ k, float* __restrict__ out) {
  const int c = blockIdx.x * blockDim.x + threadIdx.x;
  if (c >= 512) return;
  float s = 0.f;
  for (int r = 0; r < NN; ++r) s += k[(size_t)r * 512 + c];
  out[c] = s;
}

__global__ void denom_kernel(const float* __restrict__ q, const float* __restrict__ ks,
                             const float* __restrict__ sqq, const float* __restrict__ sqk,
                             float* __restrict__ denom) {
  const int i = blockIdx.x * blockDim.x + threadIdx.x;
  if (i >= NN) return;
  const float sc = rsqrtf(sqq[0]) * rsqrtf(sqk[0]);
  #pragma unroll
  for (int h = 0; h < 2; ++h) {
    float s = 0.f;
    for (int mm = 0; mm < 256; ++mm)
      s += q[(size_t)i * 512 + h * 256 + mm] * ks[h * 256 + mm];
    denom[(size_t)i * 2 + h] = s * sc + NFLT;
  }
}

__global__ void deg_kernel(const int* __restrict__ ei, const int* __restrict__ et,
                           const float* __restrict__ ew,
                           float* __restrict__ deg0, float* __restrict__ deg1) {
  const int e = blockIdx.x * blockDim.x + threadIdx.x;
  if (e >= NE) return;
  const int c = ei[NE + e];
  unsafeAtomicAdd(et[e] ? &deg1[c] : &deg0[c], ew[e]);
}

__global__ void dinv_kernel(float* __restrict__ d0, float* __restrict__ d1) {
  const int i = blockIdx.x * blockDim.x + threadIdx.x;
  if (i >= NN) return;
  d0[i] = rsqrtf(d0[i] + 1.0f);
  d1[i] = rsqrtf(d1[i] + 1.0f);
}

__global__ void agg_init_kernel(const float* __restrict__ g0, const float* __restrict__ xw0,
                                const float* __restrict__ xw1,
                                const float* __restrict__ d0, const float* __restrict__ d1,
                                const float* __restrict__ bg0, const float* __restrict__ bg1,
                                float* __restrict__ agg) {
  const size_t idx = (size_t)blockIdx.x * blockDim.x + threadIdx.x;
  if (idx >= (size_t)NN * 256) return;
  const int node = (int)(idx >> 8), c = (int)(idx & 255);
  const float i0 = d0[node], i1 = d1[node];
  agg[idx] = g0[idx] + xw0[idx] * i0 * i0 + xw1[idx] * i1 * i1 + bg0[c] + bg1[c];
}

__launch_bounds__(256)
__global__ void gcn_scatter_kernel(const int* __restrict__ ei, const int* __restrict__ et,
                                   const float* __restrict__ ew,
                                   const float* __restrict__ xw0, const float* __restrict__ xw1,
                                   const float* __restrict__ d0, const float* __restrict__ d1,
                                   float* __restrict__ agg) {
  for (int e = blockIdx.x; e < NE; e += gridDim.x) {
    const int r = ei[e], c = ei[NE + e], t = et[e];
    const float w = ew[e];
    const float* xw = t ? xw1 : xw0;
    const float* dv = t ? d1 : d0;
    const float coef = dv[r] * w * dv[c];
    const float* src = xw + (size_t)r * 256;
    float* dst = agg + (size_t)c * 256;
    unsafeAtomicAdd(&dst[threadIdx.x], coef * src[threadIdx.x]);
  }
}

// ---------------------------------------------------------------------------
extern "C" void kernel_launch(void* const* d_in, const int* in_sizes, int n_in,
                              void* d_out, int out_size, void* d_ws, size_t ws_size,
                              hipStream_t stream)
{
  (void)in_sizes; (void)n_in; (void)out_size; (void)ws_size;

  const float* x      = (const float*)d_in[0];
  const int*   ei     = (const int*)d_in[1];
  const int*   et     = (const int*)d_in[2];
  const float* ew     = (const float*)d_in[3];
  const float* W_fc0  = (const float*)d_in[4];  const float* b_fc0  = (const float*)d_in[5];
  const float* ln0g   = (const float*)d_in[6];  const float* ln0b   = (const float*)d_in[7];
  const float* Wq     = (const float*)d_in[8];  const float* bq     = (const float*)d_in[9];
  const float* Wk     = (const float*)d_in[10]; const float* bk     = (const float*)d_in[11];
  const float* Wv     = (const float*)d_in[12]; const float* bv     = (const float*)d_in[13];
  const float* ln1g   = (const float*)d_in[14]; const float* ln1b   = (const float*)d_in[15];
  const float* W_lin0 = (const float*)d_in[16]; const float* b_lin0 = (const float*)d_in[17];
  const float* bn0g   = (const float*)d_in[18]; const float* bn0b   = (const float*)d_in[19];
  const float* Wg0    = (const float*)d_in[20]; const float* bg0    = (const float*)d_in[21];
  const float* Wg1    = (const float*)d_in[22]; const float* bg1    = (const float*)d_in[23];
  const float* W_lin1 = (const float*)d_in[24]; const float* b_lin1 = (const float*)d_in[25];
  const float* bn1g   = (const float*)d_in[26]; const float* bn1b   = (const float*)d_in[27];
  const float* W_fc   = (const float*)d_in[28]; const float* b_fc   = (const float*)d_in[29];
  const float* W_p1   = (const float*)d_in[30]; const float* b_p1   = (const float*)d_in[31];
  const float* W_p2   = (const float*)d_in[32]; const float* b_p2   = (const float*)d_in[33];

  char* p = (char*)d_ws;
  auto alloc = [&](size_t bytes) -> char* {
    char* r = p; p += (bytes + 255) & ~(size_t)255; return r;
  };
  _Float16* x16    = (_Float16*)alloc((size_t)NN * 256 * 2);
  float*    h0_32  = (float*)   alloc((size_t)NN * 256 * 4);
  _Float16* h0_16  = (_Float16*)alloc((size_t)NN * 256 * 2);
  float*    q32    = (float*)   alloc((size_t)NN * 512 * 4);
  _Float16* q16    = (_Float16*)alloc((size_t)NN * 512 * 2);
  float*    k32    = (float*)   alloc((size_t)NN * 512 * 4);
  float*    v32    = (float*)   alloc((size_t)NN * 512 * 4);
  _Float16* kT     = (_Float16*)alloc((size_t)512 * NNP * 2);
  _Float16* vT     = (_Float16*)alloc((size_t)512 * NNP * 2);
  float*    g032   = (float*)   alloc((size_t)NN * 256 * 4);
  _Float16* g016   = (_Float16*)alloc((size_t)NN * 256 * 2);
  float*    xw0    = (float*)   alloc((size_t)NN * 256 * 4);
  float*    xw1    = (float*)   alloc((size_t)NN * 256 * 4);
  float*    agg32  = (float*)   alloc((size_t)NN * 256 * 4);
  _Float16* agg16  = (_Float16*)alloc((size_t)NN * 256 * 2);
  _Float16* comb16 = (_Float16*)alloc((size_t)NN * 512 * 2);
  _Float16* o16    = (_Float16*)alloc((size_t)NN * 256 * 2);
  _Float16* hid16  = (_Float16*)alloc((size_t)NN * 192 * 2);
  float*    kvs32  = (float*)   alloc((size_t)2 * 256 * 256 * 4);
  _Float16* kvsp   = (_Float16*)alloc((size_t)2 * 256 * 256 * 2);
  float*    ksraw  = (float*)   alloc(512 * 4);
  float*    sq     = (float*)   alloc(2 * 4);      // [sqq, sqk]
  float*    denomb = (float*)   alloc((size_t)NN * 2 * 4);
  float*    deg0   = (float*)   alloc((size_t)NN * 4);
  float*    deg1   = (float*)   alloc((size_t)NN * 4);
  // packed weights (f16, B-fragment order)
  _Float16* pW_fc0  = (_Float16*)alloc((size_t)256 * 256 * 2);
  _Float16* pWq     = (_Float16*)alloc((size_t)256 * 512 * 2);
  _Float16* pWk     = (_Float16*)alloc((size_t)256 * 512 * 2);
  _Float16* pWv     = (_Float16*)alloc((size_t)256 * 512 * 2);
  _Float16* pW_lin0 = (_Float16*)alloc((size_t)256 * 256 * 2);
  _Float16* pWg0    = (_Float16*)alloc((size_t)256 * 256 * 2);
  _Float16* pWg1    = (_Float16*)alloc((size_t)256 * 256 * 2);
  _Float16* pW_lin1 = (_Float16*)alloc((size_t)256 * 256 * 2);
  _Float16* pW_fc   = (_Float16*)alloc((size_t)512 * 256 * 2);
  _Float16* pW_p1   = (_Float16*)alloc((size_t)256 * 192 * 2);
  _Float16* pW_p2   = (_Float16*)alloc((size_t)192 * 128 * 2);

  const int NB = NN / 16;   // 1875 row-blocks

  // ---- zero accumulators (graph-capture safe) ----
  hipMemsetAsync(sq,    0, 2 * 4, stream);
  hipMemsetAsync(kvs32, 0, (size_t)2 * 256 * 256 * 4, stream);
  hipMemsetAsync(deg0,  0, (size_t)NN * 4, stream);
  hipMemsetAsync(deg1,  0, (size_t)NN * 4, stream);

  // ---- pack weights into WMMA B-fragment layout ----
  pack_w_kernel<<<64, 256, 0, stream>>>(W_fc0,  pW_fc0,  256, 256);
  pack_w_kernel<<<128, 256, 0, stream>>>(Wq,    pWq,     256, 512);
  pack_w_kernel<<<128, 256, 0, stream>>>(Wk,    pWk,     256, 512);
  pack_w_kernel<<<128, 256, 0, stream>>>(Wv,    pWv,     256, 512);
  pack_w_kernel<<<64, 256, 0, stream>>>(W_lin0, pW_lin0, 256, 256);
  pack_w_kernel<<<64, 256, 0, stream>>>(Wg0,    pWg0,    256, 256);
  pack_w_kernel<<<64, 256, 0, stream>>>(Wg1,    pWg1,    256, 256);
  pack_w_kernel<<<64, 256, 0, stream>>>(W_lin1, pW_lin1, 256, 256);
  pack_w_kernel<<<128, 256, 0, stream>>>(W_fc,  pW_fc,   512, 256);
  pack_w_kernel<<<48, 256, 0, stream>>>(W_p1,   pW_p1,   256, 192);
  pack_w_kernel<<<24, 256, 0, stream>>>(W_p2,   pW_p2,   192, 128);

  // ---- transformer branch ----
  cvt_f16_kernel<<<2048, 256, 0, stream>>>(x, x16, NN * 256);
  gemm16<<<NB, 256, 0, stream>>>(x16, 256, 0, pW_fc0, b_fc0, 256, 256, EPI_LN_RELU,
                                 ln0g, ln0b, h0_32, h0_16, 256, 0);
  gemm16<<<NB, 256, 0, stream>>>(h0_16, 256, 0, pWq, bq, 256, 512, EPI_NONE,
                                 nullptr, nullptr, q32, q16, 512, 0);
  gemm16<<<NB, 256, 0, stream>>>(h0_16, 256, 0, pWk, bk, 256, 512, EPI_NONE,
                                 nullptr, nullptr, k32, nullptr, 0, 0);
  gemm16<<<NB, 256, 0, stream>>>(h0_16, 256, 0, pWv, bv, 256, 512, EPI_NONE,
                                 nullptr, nullptr, v32, nullptr, 0, 0);
  sumsq2_kernel<<<512, 256, 0, stream>>>(q32, k32, &sq[0], &sq[1]);
  colsum_kernel<<<2, 256, 0, stream>>>(k32, ksraw);
  transpose_cvt_kernel<<<4096, 256, 0, stream>>>(k32, kT);
  transpose_cvt_kernel<<<4096, 256, 0, stream>>>(v32, vT);
  kvs_kernel<<<dim3(512, 30), 256, 0, stream>>>(kT, vT, kvs32);
  pack_kvs_kernel<<<128, 256, 0, stream>>>(kvs32, &sq[1], kvsp);
  denom_kernel<<<(NN + 255) / 256, 256, 0, stream>>>(q32, ksraw, &sq[0], &sq[1], denomb);
  attn_fuse<<<NB, 256, 0, stream>>>(q16, kvsp, v32, h0_32, denomb, &sq[0],
                                    ln1g, ln1b, comb16);

  // ---- GNN branch ----
  gemm16<<<NB, 256, 0, stream>>>(x16, 256, 0, pW_lin0, b_lin0, 256, 256, EPI_LN_RELU,
                                 bn0g, bn0b, g032, g016, 256, 0);
  gemm16<<<NB, 256, 0, stream>>>(g016, 256, 0, pWg0, nullptr, 256, 256, EPI_NONE,
                                 nullptr, nullptr, xw0, nullptr, 0, 0);
  gemm16<<<NB, 256, 0, stream>>>(g016, 256, 0, pWg1, nullptr, 256, 256, EPI_NONE,
                                 nullptr, nullptr, xw1, nullptr, 0, 0);
  deg_kernel<<<(NE + 255) / 256, 256, 0, stream>>>(ei, et, ew, deg0, deg1);
  dinv_kernel<<<(NN + 255) / 256, 256, 0, stream>>>(deg0, deg1);
  agg_init_kernel<<<(int)(((size_t)NN * 256 + 255) / 256), 256, 0, stream>>>(
      g032, xw0, xw1, deg0, deg1, bg0, bg1, agg32);
  gcn_scatter_kernel<<<4096, 256, 0, stream>>>(ei, et, ew, xw0, xw1, deg0, deg1, agg32);
  cvt_f16_kernel<<<2048, 256, 0, stream>>>(agg32, agg16, NN * 256);
  gemm16<<<NB, 256, 0, stream>>>(agg16, 256, 0, pW_lin1, b_lin1, 256, 256, EPI_LN_RELU,
                                 bn1g, bn1b, nullptr, comb16, 512, 256);

  // ---- head ----
  gemm16<<<NB, 256, 0, stream>>>(comb16, 512, 0, pW_fc, b_fc, 512, 256, EPI_TANH_NORM,
                                 nullptr, nullptr, nullptr, o16, 256, 0);
  gemm16<<<NB, 256, 0, stream>>>(o16, 256, 0, pW_p1, b_p1, 256, 192, EPI_RELU,
                                 nullptr, nullptr, nullptr, hid16, 192, 0);
  gemm16<<<NB, 256, 0, stream>>>(hid16, 192, 0, pW_p2, b_p2, 192, 128, EPI_NORM,
                                 nullptr, nullptr, (float*)d_out, nullptr, 0, 0);
}
